// TransformerEncoder_46986942218729
// MI455X (gfx1250) — matrix operand order repaired and checked by reading, not compiled, not run on hardware
//
#include <hip/hip_runtime.h>
#include <hip/hip_bf16.h>

// ---------------- model constants ----------------
#define BD      4
#define SEQ     1024
#define MROWS   (BD * SEQ)        // 4096
#define DIM     768
#define HEADS   12
#define DHEAD   64
#define INNER   768               // HEADS*DHEAD
#define QKV3    (3 * INNER)       // 2304
#define MLP     3072
#define DEPTH   6

typedef __attribute__((ext_vector_type(16))) _Float16 v16h;
typedef __attribute__((ext_vector_type(8)))  _Float16 v8h;
typedef __attribute__((ext_vector_type(8)))  float    v8f;

// ---------------- WMMA helpers (gfx1250, wave32) ----------------
__device__ __forceinline__ v8f wmma_f16(v16h a, v16h b, v8f c) {
    // D = A(16x32 f16) * B(32x16 f16) + C(16x16 f32)
    return __builtin_amdgcn_wmma_f32_16x16x32_f16(
        false, a, false, b, (short)0, c, false, false);
}

// A fragment 16x32 f16: lane holds row m=lane&15, half h=lane>>4.
// VGPR halves [0..7]  = K = h*8    + 0..7  (contiguous in memory)
// VGPR halves [8..15] = K = 16+h*8 + 0..7  (contiguous in memory)
// A points at element (row0,k0) of a row-major [..][lda] matrix.
__device__ __forceinline__ v16h load_a16x32(const _Float16* __restrict__ A,
                                            int lda, int lane) {
    const int m = lane & 15, h = lane >> 4;
    const _Float16* p = A + (size_t)m * lda + h * 8;
    v8h lo = *reinterpret_cast<const v8h*>(p);
    v8h hi = *reinterpret_cast<const v8h*>(p + 16);
    v16h r;
#pragma unroll
    for (int i = 0; i < 8; ++i) { r[i] = lo[i]; r[i + 8] = hi[i]; }
    return r;
}

// B fragment 32x16 f16: lane holds column n=lane&15, half h=lane>>4,
// halves [0..7] = K = h*8 + 0..7, halves [8..15] = K = 16+h*8 + 0..7 (B^T stored [n][k]).
// BT points at element (n0,k0) of B^T stored row-major [N][ldb].
__device__ __forceinline__ v16h load_b32x16(const _Float16* __restrict__ BT,
                                            int ldb, int lane) {
    const int n = lane & 15, h = lane >> 4;
    const _Float16* p = BT + (size_t)n * ldb + h * 8;
    v8h lo = *reinterpret_cast<const v8h*>(p);
    v8h hi = *reinterpret_cast<const v8h*>(p + 16);
    v16h r;
#pragma unroll
    for (int i = 0; i < 8; ++i) { r[i] = lo[i]; r[i + 8] = hi[i]; }
    return r;
}

// ---------------- weight transpose + f32->f16 convert ----------------
// in: [K][N] f32 (einsum [in,out]) -> out: [N][K] f16 (B^T layout)
__global__ void tconv_f16(const float* __restrict__ in,
                          _Float16* __restrict__ out, int K, int N) {
    size_t idx = (size_t)blockIdx.x * blockDim.x + threadIdx.x;
    size_t total = (size_t)K * N;
    if (idx >= total) return;
    int n = (int)(idx / K), k = (int)(idx % K);
    out[idx] = (_Float16)in[(size_t)k * N + n];
}

// ---------------- LayerNorm (768) -> f16 ----------------
__global__ __launch_bounds__(256) void layernorm_f16(
    const float* __restrict__ x, const float* __restrict__ w,
    const float* __restrict__ b, _Float16* __restrict__ out) {
    __shared__ float red[256];
    const int row = blockIdx.x, t = threadIdx.x;
    const float* xr = x + (size_t)row * DIM;
    float v0 = xr[t], v1 = xr[t + 256], v2 = xr[t + 512];
    red[t] = v0 + v1 + v2;
    __syncthreads();
    for (int off = 128; off > 0; off >>= 1) {
        if (t < off) red[t] += red[t + off];
        __syncthreads();
    }
    float mu = red[0] * (1.0f / DIM);
    __syncthreads();
    float d0 = v0 - mu, d1 = v1 - mu, d2 = v2 - mu;
    red[t] = d0 * d0 + d1 * d1 + d2 * d2;
    __syncthreads();
    for (int off = 128; off > 0; off >>= 1) {
        if (t < off) red[t] += red[t + off];
        __syncthreads();
    }
    float rs = rsqrtf(red[0] * (1.0f / DIM) + 1e-5f);
    _Float16* orow = out + (size_t)row * DIM;
    orow[t]       = (_Float16)(d0 * rs * w[t]       + b[t]);
    orow[t + 256] = (_Float16)(d1 * rs * w[t + 256] + b[t + 256]);
    orow[t + 512] = (_Float16)(d2 * rs * w[t + 512] + b[t + 512]);
}

// ---------------- generic WMMA GEMM with fused epilogue ----------------
// C[M][N] = A[M][K](f16) * B^T[N][K](f16)  (+bias)(+GELU)(+residual)
// Block: 256 threads = 8 waves (2 in M x 4 in N); block tile 128(M) x 256(N);
// wave tile 64x64 = 4x4 WMMA tiles (128 accumulator VGPRs) -> 16 WMMAs per
// 16 b128 loads per K-step, so the loop is WMMA-issue-bound, not L1-bound.
__global__ __launch_bounds__(256) void gemm_wmma(
    const _Float16* __restrict__ A, const _Float16* __restrict__ BT,
    const float* __restrict__ bias, const float* __restrict__ resid,
    float* __restrict__ outf, _Float16* __restrict__ outh,
    int N, int K, int gelu) {
    const int lane = threadIdx.x & 31, wave = threadIdx.x >> 5;
    const int m0 = blockIdx.y * 128 + (wave & 1) * 64;
    const int n0 = blockIdx.x * 256 + (wave >> 1) * 64;

    v8f acc[4][4];
#pragma unroll
    for (int i = 0; i < 4; ++i)
#pragma unroll
        for (int j = 0; j < 4; ++j) acc[i][j] = {};

    const int lm = lane & 15;
    for (int k = 0; k < K; k += 32) {
        // prefetch two K-steps ahead, WGP scope (pull into all cache levels)
        if (k + 64 < K) {
            __builtin_prefetch(A + (size_t)(m0 + lm) * K + k + 64, 0, 3);
            __builtin_prefetch(BT + (size_t)(n0 + lm) * K + k + 64, 0, 3);
        }
        v16h a[4], b[4];
#pragma unroll
        for (int i = 0; i < 4; ++i)
            a[i] = load_a16x32(A + (size_t)(m0 + 16 * i) * K + k, K, lane);
#pragma unroll
        for (int j = 0; j < 4; ++j)
            b[j] = load_b32x16(BT + (size_t)(n0 + 16 * j) * K + k, K, lane);
#pragma unroll
        for (int i = 0; i < 4; ++i)
#pragma unroll
            for (int j = 0; j < 4; ++j)
                acc[i][j] = wmma_f16(a[i], b[j], acc[i][j]);
    }

    // C layout: VGPR r -> row (r + 8*(lane>=16)), col = lane&15
    const int lc = lane & 15, lh = lane >> 4;
#pragma unroll
    for (int mi = 0; mi < 4; ++mi) {
#pragma unroll
        for (int ni = 0; ni < 4; ++ni) {
            const int colg = n0 + ni * 16 + lc;
            const float bv = bias ? bias[colg] : 0.0f;
#pragma unroll
            for (int r = 0; r < 8; ++r) {
                const int rowg = m0 + mi * 16 + r + lh * 8;
                const size_t o = (size_t)rowg * N + colg;
                float v = acc[mi][ni][r] + bv;
                if (gelu) v = 0.5f * v * (1.0f + erff(v * 0.70710678f));
                if (resid) v += resid[o];
                if (outf) outf[o] = v;
                if (outh) outh[o] = (_Float16)v;
            }
        }
    }
}

// ---------------- flash attention (f16 WMMA, online softmax) ----------------
// grid: BD*HEADS*(SEQ/128) blocks, 256 threads (8 waves, 16 q-rows per wave)
__global__ __launch_bounds__(256) void flash_attn(
    const _Float16* __restrict__ qkv,   // [MROWS][QKV3]
    _Float16* __restrict__ attn_out) {  // [MROWS][INNER]
    __shared__ _Float16 ldsV[DHEAD * 32];      // V^T: [dim][key]   (4 KB)
    __shared__ _Float16 ldsP[8 * 16 * 32];     // per-wave P [16][32] (8 KB)
    const int lane = threadIdx.x & 31, wave = threadIdx.x >> 5;
    const int chunk = blockIdx.x & 7;
    const int h = (blockIdx.x >> 3) % HEADS;
    const int b = blockIdx.x / (8 * HEADS);
    const int qs = chunk * 128 + wave * 16;      // q-row base of this wave
    const size_t rowbase = (size_t)(b * SEQ);

    // Q fragments (A operand, 16 rows x K=64 as two 16x32 frags)
    const _Float16* Qb = qkv + (rowbase + qs) * QKV3 + h * DHEAD;
    const v16h qa0 = load_a16x32(Qb, QKV3, lane);
    const v16h qa1 = load_a16x32(Qb + 32, QKV3, lane);

    float mrow[8], lrow[8];
    v8f o0 = {}, o1 = {}, o2 = {}, o3 = {};
#pragma unroll
    for (int r = 0; r < 8; ++r) { mrow[r] = -1e30f; lrow[r] = 0.0f; }

    const int lc = lane & 15, lh = lane >> 4;
    _Float16* myP = ldsP + wave * (16 * 32);

    for (int kc = 0; kc < SEQ; kc += 32) {
        __syncthreads();   // protect ldsV reads from previous iteration
        // stage V^T chunk: 32 keys x 64 dims -> ldsV[dim][key]
        for (int i = threadIdx.x; i < 32 * DHEAD; i += 256) {
            const int key = i & 31, dim = i >> 5;
            ldsV[dim * 32 + key] =
                qkv[(rowbase + kc + key) * QKV3 + 2 * INNER + h * DHEAD + dim];
        }
        __syncthreads();

        // scores: S(16x32) = Q(16x64) * K^T ; K stored row-major [key][dim]
        const _Float16* Kb = qkv + (rowbase + kc) * QKV3 + INNER + h * DHEAD;
        v8f s0 = {}, s1 = {};
        s0 = wmma_f16(qa0, load_b32x16(Kb, QKV3, lane), s0);
        s0 = wmma_f16(qa1, load_b32x16(Kb + 32, QKV3, lane), s0);
        s1 = wmma_f16(qa0, load_b32x16(Kb + (size_t)16 * QKV3, QKV3, lane), s1);
        s1 = wmma_f16(qa1, load_b32x16(Kb + (size_t)16 * QKV3 + 32, QKV3, lane), s1);
        s0 *= 0.125f;  // 1/sqrt(64)
        s1 *= 0.125f;

        // online softmax; row r (+8 for upper half) spans 16 lanes of a half
        float p0[8], p1[8];
#pragma unroll
        for (int r = 0; r < 8; ++r) {
            float mx = fmaxf(s0[r], s1[r]);
#pragma unroll
            for (int off = 1; off < 16; off <<= 1)
                mx = fmaxf(mx, __shfl_xor(mx, off, 32));
            const float mnew = fmaxf(mrow[r], mx);
            const float fac = __expf(mrow[r] - mnew);
            const float e0 = __expf(s0[r] - mnew);
            const float e1 = __expf(s1[r] - mnew);
            float rs = e0 + e1;
#pragma unroll
            for (int off = 1; off < 16; off <<= 1)
                rs += __shfl_xor(rs, off, 32);
            lrow[r] = lrow[r] * fac + rs;
            mrow[r] = mnew;
            o0[r] *= fac; o1[r] *= fac; o2[r] *= fac; o3[r] *= fac;
            p0[r] = e0; p1[r] = e1;
        }

        // C-layout -> A-layout transpose of P through LDS
#pragma unroll
        for (int r = 0; r < 8; ++r) {
            const int prow = r + lh * 8;
            myP[prow * 32 + lc]      = (_Float16)p0[r];
            myP[prow * 32 + 16 + lc] = (_Float16)p1[r];
        }
        __syncthreads();
        const v16h pa = load_a16x32(myP, 32, lane);

        // O += P(16x32) * V(32x64): B operand from ldsV (V^T [dim][key])
        o0 = wmma_f16(pa, load_b32x16(ldsV +  0 * 32, 32, lane), o0);
        o1 = wmma_f16(pa, load_b32x16(ldsV + 16 * 32, 32, lane), o1);
        o2 = wmma_f16(pa, load_b32x16(ldsV + 32 * 32, 32, lane), o2);
        o3 = wmma_f16(pa, load_b32x16(ldsV + 48 * 32, 32, lane), o3);
    }

    // normalize and write out f16 [MROWS][INNER]
#pragma unroll
    for (int r = 0; r < 8; ++r) {
        const float inv = 1.0f / lrow[r];
        const size_t rowg = rowbase + qs + r + lh * 8;
        _Float16* orow = attn_out + rowg * INNER + h * DHEAD + lc;
        orow[0]  = (_Float16)(o0[r] * inv);
        orow[16] = (_Float16)(o1[r] * inv);
        orow[32] = (_Float16)(o2[r] * inv);
        orow[48] = (_Float16)(o3[r] * inv);
    }
}

// ---------------- host-side orchestration ----------------
extern "C" void kernel_launch(void* const* d_in, const int* in_sizes, int n_in,
                              void* d_out, int out_size, void* d_ws, size_t ws_size,
                              hipStream_t stream) {
    (void)in_sizes; (void)n_in; (void)out_size; (void)ws_size;
    const float* in_x    = (const float*)d_in[0];
    const float* ln1_w   = (const float*)d_in[1];
    const float* ln1_b   = (const float*)d_in[2];
    const float* Wqkv    = (const float*)d_in[3];
    const float* Wo      = (const float*)d_in[4];
    const float* bo      = (const float*)d_in[5];
    const float* ln2_w   = (const float*)d_in[6];
    const float* ln2_b   = (const float*)d_in[7];
    const float* W1      = (const float*)d_in[8];
    const float* b1      = (const float*)d_in[9];
    const float* W2      = (const float*)d_in[10];
    const float* b2      = (const float*)d_in[11];

    // workspace layout (all offsets 256B-aligned)
    char* ws = (char*)d_ws;
    constexpr size_t SZ_X    = (size_t)MROWS * DIM * 4;      // 12.6 MB f32
    constexpr size_t SZ_LN   = (size_t)MROWS * DIM * 2;      // f16
    constexpr size_t SZ_QKV  = (size_t)MROWS * QKV3 * 2;     // f16
    constexpr size_t SZ_ATT  = (size_t)MROWS * INNER * 2;    // f16
    constexpr size_t SZ_H1   = (size_t)MROWS * MLP * 2;      // f16
    constexpr size_t SZ_WQKV = (size_t)DIM * QKV3 * 2;
    constexpr size_t SZ_WO   = (size_t)INNER * DIM * 2;
    constexpr size_t SZ_W1   = (size_t)DIM * MLP * 2;
    constexpr size_t SZ_W2   = (size_t)MLP * DIM * 2;
    size_t off = 0;
    float*     X     = (float*)(ws + off);     off += SZ_X;
    float*     Y     = (float*)(ws + off);     off += SZ_X;
    _Float16*  LN    = (_Float16*)(ws + off);  off += SZ_LN;
    _Float16*  QKVb  = (_Float16*)(ws + off);  off += SZ_QKV;
    _Float16*  ATT   = (_Float16*)(ws + off);  off += SZ_ATT;
    _Float16*  H1    = (_Float16*)(ws + off);  off += SZ_H1;
    _Float16*  WTqkv = (_Float16*)(ws + off);  off += SZ_WQKV;
    _Float16*  WTo   = (_Float16*)(ws + off);  off += SZ_WO;
    _Float16*  WT1   = (_Float16*)(ws + off);  off += SZ_W1;
    _Float16*  WT2   = (_Float16*)(ws + off);  off += SZ_W2;

    hipMemcpyAsync(X, in_x, SZ_X, hipMemcpyDeviceToDevice, stream);

    const dim3 blk(256);
    const dim3 gLN(MROWS);
    const dim3 gAttn(BD * HEADS * (SEQ / 128));
    auto cdiv = [](size_t a, size_t b) { return (unsigned)((a + b - 1) / b); };

    for (int l = 0; l < DEPTH; ++l) {
        // per-layer weight transpose+convert (f32 [in][out] -> f16 [out][in])
        tconv_f16<<<cdiv((size_t)DIM * QKV3, 256), blk, 0, stream>>>(
            Wqkv + (size_t)l * DIM * QKV3, WTqkv, DIM, QKV3);
        tconv_f16<<<cdiv((size_t)INNER * DIM, 256), blk, 0, stream>>>(
            Wo + (size_t)l * INNER * DIM, WTo, INNER, DIM);
        tconv_f16<<<cdiv((size_t)DIM * MLP, 256), blk, 0, stream>>>(
            W1 + (size_t)l * DIM * MLP, WT1, DIM, MLP);
        tconv_f16<<<cdiv((size_t)MLP * DIM, 256), blk, 0, stream>>>(
            W2 + (size_t)l * MLP * DIM, WT2, MLP, DIM);

        // ---- attention block ----
        layernorm_f16<<<gLN, blk, 0, stream>>>(
            X, ln1_w + (size_t)l * DIM, ln1_b + (size_t)l * DIM, LN);
        gemm_wmma<<<dim3(QKV3 / 256, MROWS / 128), blk, 0, stream>>>(
            LN, WTqkv, nullptr, nullptr, nullptr, QKVb, QKV3, DIM, 0);
        flash_attn<<<gAttn, blk, 0, stream>>>(QKVb, ATT);
        gemm_wmma<<<dim3(DIM / 256, MROWS / 128), blk, 0, stream>>>(
            ATT, WTo, bo + (size_t)l * DIM, X, Y, nullptr, DIM, INNER, 0);

        // ---- MLP block ----
        layernorm_f16<<<gLN, blk, 0, stream>>>(
            Y, ln2_w + (size_t)l * DIM, ln2_b + (size_t)l * DIM, LN);
        gemm_wmma<<<dim3(MLP / 256, MROWS / 128), blk, 0, stream>>>(
            LN, WT1, b1 + (size_t)l * MLP, nullptr, nullptr, H1, MLP, DIM, 1);
        gemm_wmma<<<dim3(DIM / 256, MROWS / 128), blk, 0, stream>>>(
            H1, WT2, b2 + (size_t)l * DIM, Y, X, nullptr, DIM, MLP, 0);
    }

    hipMemcpyAsync(d_out, X, SZ_X, hipMemcpyDeviceToDevice, stream);
}
